// S4DKernel_67216238183019
// MI455X (gfx1250) — matrix operand order repaired and checked by reading, not compiled, not run on hardware
//
#include <hip/hip_runtime.h>
#include <hip/hip_bf16.h>
#include <math.h>

// S4D kernel generation on gfx1250 (MI455X).
//
// K[h,l] = 2*Re( sum_n Cm[h,n] * exp(dtA[h,n]*l) ),  H=1024, N/2=32, L=4096.
//
// Decompose l = l1 + 16*l2 -> per-head complex GEMM (16x32)x(32x256):
//   K = Re(U*V) = Ur*Vr + (-Ui)*Vi   (two v_wmma_f32_16x16x32_f16 per tile)
// U  = 2*Cm[n]*z_n^l1            (A matrix, f16)
// V  = (z_n^16)^l2               (B panel; f32 geometric recurrence across
//                                 tiles, step factor z_n^256, rounded to f16)
// One wave per head; 16 tiles of 16x16 f32 accumulators per head.
// Prologue uses hardware transcendentals (v_exp/v_sin/v_cos_f32) so setup
// stays below the 23.3 TB/s store floor for the 16.8 MB output.

typedef __attribute__((ext_vector_type(16))) _Float16 v16h;
typedef __attribute__((ext_vector_type(8)))  float    v8f;
typedef __attribute__((ext_vector_type(4)))  float    v4f;

#define S4D_H 1024
#define S4D_N 32     // N/2 modes
#define S4D_L 4096
#define S4D_T 16     // tiles of 16 l2-columns (16*16*16 = 4096 l values)

__global__ __launch_bounds__(256) void s4d_wmma_kernel(
    const float* __restrict__ C,          // (H, 32, 2) interleaved re/im
    const float* __restrict__ log_dt,     // (H)
    const float* __restrict__ log_A_real, // (H, 32)
    const float* __restrict__ A_imag,     // (H, 32)
    float* __restrict__ out)              // (H, L)
{
    const int lane = threadIdx.x & 31;
    const int wave = threadIdx.x >> 5;
    const int h    = (blockIdx.x << 3) + wave;   // one wave per head
    if (h >= S4D_H) return;                      // wave-uniform, never taken

    const int mrow   = lane & 15;          // A-matrix row M = l1
    const int ncol   = lane & 15;          // B/D column N (= l2 offset)
    const int kbaseA = (lane >> 4) << 3;   // A K-group base: 0 or 8
    const int kbaseB = (lane >> 4) << 4;   // B K-group base: 0 or 16

    const float dt = __expf(log_dt[h]);

    v16h Ar, Ain;                 // A real, A (negated) imag, f16
    float vr[16], vi[16];         // B panel state (f32 recurrence)
    float sr[16], si[16];         // per-element step factor z_n^256

    #pragma unroll
    for (int e = 0; e < 16; ++e) {
        // ---- A matrix element: (M = mrow, K = nA) per ISA 16-bit A layout
        const int nA = kbaseA + e + ((e & 8) ? 8 : 0);
        {
            const float ar   = -__expf(log_A_real[h * S4D_N + nA]);
            const float ai   = A_imag[h * S4D_N + nA];
            const float dtar = dt * ar;
            const float dtai = dt * ai;
            // E = exp(dtA)
            const float em = __expf(dtar);
            const float Er = em * __cosf(dtai);
            const float Ei = em * __sinf(dtai);
            // g = (E - 1)/A  via conj(A)/|A|^2
            const float inv = __frcp_rn(ar * ar + ai * ai);
            const float gr  = ((Er - 1.0f) * ar + Ei * ai) * inv;
            const float gi  = (Ei * ar - (Er - 1.0f) * ai) * inv;
            // Cm2 = 2 * C * g
            const float cr  = C[(h * S4D_N + nA) * 2 + 0];
            const float ci  = C[(h * S4D_N + nA) * 2 + 1];
            const float cmr = 2.0f * (cr * gr - ci * gi);
            const float cmi = 2.0f * (cr * gi + ci * gr);
            // U = Cm2 * z^mrow,  z^mrow = exp(dtA * mrow)
            const float pm = __expf(dtar * (float)mrow);
            const float pc = __cosf(dtai * (float)mrow);
            const float ps = __sinf(dtai * (float)mrow);
            const float ur = pm * (cmr * pc - cmi * ps);
            const float ui = pm * (cmr * ps + cmi * pc);
            Ar[e]  = (_Float16)ur;
            Ain[e] = (_Float16)(-ui);     // pre-negated: Re = Ur*Vr + (-Ui)*Vi
        }
        // ---- B panel seed + step: (K = nB, N = ncol) per ISA 16-bit B layout
        {
            const int   nB   = kbaseA + e + ((e & 8) ? 8 : 0);
            const float ar   = -__expf(log_A_real[h * S4D_N + nB]);
            const float ai   = A_imag[h * S4D_N + nB];
            const float dtar = dt * ar;
            const float dtai = dt * ai;
            // seed: z^(16*ncol)
            const float m0 = __expf(dtar * (16.0f * (float)ncol));
            vr[e] = m0 * __cosf(dtai * (16.0f * (float)ncol));
            vi[e] = m0 * __sinf(dtai * (16.0f * (float)ncol));
            // step per tile: z^256
            const float ms = __expf(dtar * 256.0f);
            sr[e] = ms * __cosf(dtai * 256.0f);
            si[e] = ms * __sinf(dtai * 256.0f);
        }
    }

    float* __restrict__ outh = out + (size_t)h * S4D_L;
    const int mhi = (lane >> 4) << 3;     // D-matrix: VGPR r -> M = r + mhi

    #pragma unroll 1
    for (int t = 0; t < S4D_T; ++t) {
        // round B panel to f16 for the matrix pipe
        v16h Br, Bi;
        #pragma unroll
        for (int e = 0; e < 16; ++e) {
            Br[e] = (_Float16)vr[e];
            Bi[e] = (_Float16)vi[e];
        }

        v8f acc = {};
        acc = __builtin_amdgcn_wmma_f32_16x16x32_f16(
                  false, Ar,  false, Br, (short)0, acc, false, false);
        acc = __builtin_amdgcn_wmma_f32_16x16x32_f16(
                  false, Ain, false, Bi, (short)0, acc, false, false);

        // D[r] -> l = (r + mhi) + 16*(16*t + ncol): 8 contiguous floats/lane
        const int base = mhi + 16 * (16 * t + ncol);
        v4f lo = { acc[0], acc[1], acc[2], acc[3] };
        v4f hi = { acc[4], acc[5], acc[6], acc[7] };
        *(v4f*)(outh + base)     = lo;
        *(v4f*)(outh + base + 4) = hi;

        // advance B panel one tile: multiply by z^256 (f32 complex mul)
        #pragma unroll
        for (int e = 0; e < 16; ++e) {
            const float nr = vr[e] * sr[e] - vi[e] * si[e];
            const float ni = vr[e] * si[e] + vi[e] * sr[e];
            vr[e] = nr;
            vi[e] = ni;
        }
    }
}

extern "C" void kernel_launch(void* const* d_in, const int* in_sizes, int n_in,
                              void* d_out, int out_size, void* d_ws, size_t ws_size,
                              hipStream_t stream) {
    (void)in_sizes; (void)n_in; (void)d_ws; (void)ws_size; (void)out_size;
    const float* C          = (const float*)d_in[0];  // (H, 32, 2)
    const float* log_dt     = (const float*)d_in[1];  // (H)
    const float* log_A_real = (const float*)d_in[2];  // (H, 32)
    const float* A_imag     = (const float*)d_in[3];  // (H, 32)
    float* out              = (float*)d_out;          // (H, L)

    // 8 waves per block, one head per wave -> 1024/8 = 128 blocks
    s4d_wmma_kernel<<<dim3(S4D_H / 8), dim3(256), 0, stream>>>(
        C, log_dt, log_A_real, A_imag, out);
}
